// AttentionalSeq2seq_80693845557252
// MI455X (gfx1250) — matrix operand order repaired and checked by reading, not compiled, not run on hardware
//
#include <hip/hip_runtime.h>
#include <hip/hip_bf16.h>

#define H 1024
#define B 64
#define S_IN 1024
#define S_OUT 256
#define OUT_N 512
#define H3 3072

#if defined(__has_builtin)
#if __has_builtin(__builtin_amdgcn_sched_barrier)
#define SCHED_FENCE() __builtin_amdgcn_sched_barrier(0)
#endif
#endif
#ifndef SCHED_FENCE
#define SCHED_FENCE()
#endif

typedef unsigned int u32;
typedef unsigned short u16;
typedef __attribute__((ext_vector_type(16))) __bf16 v16bf;
typedef __attribute__((ext_vector_type(8))) float v8f;
typedef __attribute__((ext_vector_type(4))) u32 u32x4;

union ABReg { v16bf v; u32x4 q[2]; };

__device__ __forceinline__ u16 f2bf(float f) {
  u32 u = __float_as_uint(f);
  u32 r = u + 0x7FFFu + ((u >> 16) & 1u);   // round-to-nearest-even
  return (u16)(r >> 16);
}
__device__ __forceinline__ float bf2f(u16 h) { return __uint_as_float(((u32)h) << 16); }
__device__ __forceinline__ float sigmoidf(float x) { return 1.f / (1.f + __expf(-x)); }

// ---------------------------------------------------------------------------
// Software-pipelined WMMA GEMM core.
//   NT     : 16-wide N tiles sharing one A tile (independent acc chains)
//   KTILES : K/32 (fully unrolled)
//   DEPTH  : register pipeline stages.  sched_barrier(0) fences keep the
//            compiler from sinking prefetch loads down to their use, so the
//            WMMA for slot k waits only partially while slots k+1..k+DEPTH-1
//            are in flight.
// A: MxK bf16 row-major (lda), W: NxK bf16 row-major (ldw), f32 accumulate.
// Lane layout (wave32): sub = lane&15, hi = lane>>4.
//   A operand: row m0+sub; elems 0..7 = K[k0+8*hi ..], 8..15 = K[k0+16+8*hi ..]
//   B operand: W row n+sub; elems 0..7 = K[k0+8*hi ..], 8..15 = K[k0+16+8*hi ..]
// ---------------------------------------------------------------------------
template <int NT, int KTILES, int DEPTH>
__device__ __forceinline__ void wmma_dot_pipe(const u16* A, int lda, const u16* W, int ldw,
                                              int m0, int n0, v8f acc[NT]) {
  const int lane = threadIdx.x & 31;
  const int sub  = lane & 15;
  const int hi   = lane >> 4;
  const u16* arow = A + (size_t)(m0 + sub) * lda;
  const u16* wrow = W + (size_t)(n0 + sub) * ldw;

  ABReg a[DEPTH];
  ABReg b[DEPTH][NT];
  auto load_stage = [&](int d, int k0) {
    a[d].q[0] = *(const u32x4*)(arow + k0 + hi * 8);
    a[d].q[1] = *(const u32x4*)(arow + k0 + 16 + hi * 8);
#pragma unroll
    for (int i = 0; i < NT; ++i) {
      const u16* wr = wrow + (size_t)i * 16 * ldw;
      b[d][i].q[0] = *(const u32x4*)(wr + k0 + hi * 8);
      b[d][i].q[1] = *(const u32x4*)(wr + k0 + 16 + hi * 8);
    }
  };

#pragma unroll
  for (int d = 0; d < DEPTH && d < KTILES; ++d) load_stage(d, d * 32);
  SCHED_FENCE();

#pragma unroll
  for (int kk = 0; kk < KTILES; ++kk) {
    const int cur = kk % DEPTH;
#pragma unroll
    for (int i = 0; i < NT; ++i)
      acc[i] = __builtin_amdgcn_wmma_f32_16x16x32_bf16(false, a[cur].v, false, b[cur][i].v,
                                                       (short)0, acc[i], false, false);
    SCHED_FENCE();                       // WMMAs for slot kk stay above...
    if (kk + DEPTH < KTILES) load_stage(cur, (kk + DEPTH) * 32);
    SCHED_FENCE();                       // ...and slot-reuse prefetch stays here
  }
}

// C/D tile layout: element (m,n) lives in lane (n&15) + 16*(m>=8), vgpr m&7.
__device__ __forceinline__ void store_tile(float* C, int ldc, int m0, int n0, v8f acc,
                                           const float* bias) {
  const int lane = threadIdx.x & 31, sub = lane & 15, hi = lane >> 4;
  float bv = bias ? bias[n0 + sub] : 0.f;
#pragma unroll
  for (int v = 0; v < 8; ++v)
    C[(size_t)(m0 + v + 8 * hi) * ldc + n0 + sub] = acc[v] + bv;
}

__device__ __forceinline__ void store_tile_dual(float* Cf, u16* Cb, int ldc, int m0, int n0,
                                                v8f acc, const float* bias) {
  const int lane = threadIdx.x & 31, sub = lane & 15, hi = lane >> 4;
  float bv = bias[n0 + sub];
#pragma unroll
  for (int v = 0; v < 8; ++v) {
    float val = acc[v] + bv;
    size_t idx = (size_t)(m0 + v + 8 * hi) * ldc + n0 + sub;
    Cf[idx] = val;
    Cb[idx] = f2bf(val);
  }
}

// ---------------------------------------------------------------------------
// Device-wide generation barrier (persistent-kernel grid sync).
// bar[0] = arrival count, bar[1] = generation.
// ---------------------------------------------------------------------------
__device__ __forceinline__ void grid_sync(u32* bar, u32 nblk) {
  __threadfence();            // release my stores to device scope
  __syncthreads();
  if (threadIdx.x == 0) {
    u32 gen  = __hip_atomic_load(&bar[1], __ATOMIC_RELAXED, __HIP_MEMORY_SCOPE_AGENT);
    u32 prev = __hip_atomic_fetch_add(&bar[0], 1u, __ATOMIC_ACQ_REL, __HIP_MEMORY_SCOPE_AGENT);
    if (prev == nblk - 1) {
      __hip_atomic_store(&bar[0], 0u, __ATOMIC_RELAXED, __HIP_MEMORY_SCOPE_AGENT);
      __hip_atomic_fetch_add(&bar[1], 1u, __ATOMIC_RELEASE, __HIP_MEMORY_SCOPE_AGENT);
    } else {
      while (__hip_atomic_load(&bar[1], __ATOMIC_ACQUIRE, __HIP_MEMORY_SCOPE_AGENT) == gen)
        __builtin_amdgcn_s_sleep(4);
    }
  }
  __syncthreads();
  __threadfence();            // acquire: invalidate stale near caches
}

// ---------------------------------------------------------------------------
// Utility kernels
// ---------------------------------------------------------------------------
__global__ void cvt_kernel(const float* __restrict__ src, u16* __restrict__ dst, int n) {
  int stride = gridDim.x * blockDim.x;
  for (int i = blockIdx.x * blockDim.x + threadIdx.x; i < n; i += stride)
    dst[i] = f2bf(src[i]);
}

__global__ void init_kernel(u32* bar, float* h_f, u16* h_bf) {
  int i = blockIdx.x * blockDim.x + threadIdx.x;
  if (i < 8) bar[i] = 0;
  int stride = gridDim.x * blockDim.x;
  for (int j = i; j < B * H; j += stride) { h_f[j] = 0.f; h_bf[j] = (u16)0; }
}

// ---------------------------------------------------------------------------
// Persistent encoder: 1024 GRU steps.  Per step:
//   gh = h @ enc_W_hh^T  (WMMA, 4 Mtiles x 96 Ngroups(NT=2) = 384 wave-groups,
//   exactly one per wave at 48 blocks x 8 waves)        -> grid_sync
//   gates + h update + write enc_out (bf16)             -> grid_sync
// ---------------------------------------------------------------------------
__global__ void encoder_kernel(const float* __restrict__ x, const float* __restrict__ Wih,
                               const float* __restrict__ bih, const float* __restrict__ bhh,
                               const u16* __restrict__ Whh_bf, u16* h_bf, float* h_f,
                               float* gh, u16* enc_bf, u32* bar, int nblk) {
  const int tid = blockIdx.x * blockDim.x + threadIdx.x;
  const int nthreads = nblk * blockDim.x;
  const int wave = tid >> 5;
  const int nwaves = nthreads >> 5;
  const int NG = (B / 16) * (H3 / 32);    // 384 groups of 1 Mtile x 2 Ntiles

  for (int t = 0; t < S_IN; ++t) {
    for (int g = wave; g < NG; g += nwaves) {
      int tm = (g & 3) * 16;
      int tn = (g >> 2) * 32;
      v8f acc[2] = {{}, {}};
      wmma_dot_pipe<2, H / 32, 4>(h_bf, H, Whh_bf, H, tm, tn, acc);
      store_tile(gh, H3, tm, tn, acc[0], nullptr);
      store_tile(gh, H3, tm, tn + 16, acc[1], nullptr);
    }
    grid_sync(bar, nblk);

    for (int i = tid; i < B * H; i += nthreads) {
      int b = i >> 10, j = i & (H - 1);
      float xv = x[b * S_IN + t];
      float ir = xv * Wih[j]          + bih[j];
      float iz = xv * Wih[H + j]      + bih[H + j];
      float in_ = xv * Wih[2 * H + j] + bih[2 * H + j];
      const float* g = gh + (size_t)b * H3;
      float gr = g[j]         + bhh[j];
      float gz = g[H + j]     + bhh[H + j];
      float gn = g[2 * H + j] + bhh[2 * H + j];
      float r = sigmoidf(ir + gr);
      float z = sigmoidf(iz + gz);
      float n = tanhf(in_ + r * gn);
      float hv = (1.f - z) * n + z * h_f[i];
      h_f[i] = hv;
      u16 hb = f2bf(hv);
      h_bf[i] = hb;
      enc_bf[((size_t)b * S_IN + t) * H + j] = hb;
    }
    grid_sync(bar, nblk);
  }
}

// ---------------------------------------------------------------------------
// Attention energies: E[r] = sum_n X[r,n] * ((X @ attn_W^T)[r,n] + attn_b[n]),
// X = enc_out viewed as (B*S_IN, H) bf16.  One 16-row tile per wave iteration;
// N blocked 4-wide per A load (NT=4), 3-deep pipeline.
// ---------------------------------------------------------------------------
__global__ void attn_energy_kernel(const u16* __restrict__ X, const u16* __restrict__ Wa,
                                   const float* __restrict__ ba, float* __restrict__ energies) {
  const int lane = threadIdx.x & 31, sub = lane & 15, hi = lane >> 4;
  const int wave = (blockIdx.x * blockDim.x + threadIdx.x) >> 5;
  const int nwaves = (gridDim.x * blockDim.x) >> 5;
  const int NTILE = (B * S_IN) / 16;      // 4096 row tiles

  for (int tile = wave; tile < NTILE; tile += nwaves) {
    int r0 = tile * 16;
    float ep[8] = {0.f, 0.f, 0.f, 0.f, 0.f, 0.f, 0.f, 0.f};
    for (int n0 = 0; n0 < H; n0 += 64) {
      v8f acc[4] = {{}, {}, {}, {}};
      wmma_dot_pipe<4, H / 32, 3>(X, H, Wa, H, r0, n0, acc);
#pragma unroll
      for (int gidx = 0; gidx < 4; ++gidx) {
        int n = n0 + gidx * 16 + sub;
        float bn = ba[n];
#pragma unroll
        for (int v = 0; v < 8; ++v) {
          float xv = bf2f(X[(size_t)(r0 + v + 8 * hi) * H + n]);
          ep[v] += xv * (acc[gidx][v] + bn);
        }
      }
    }
    // reduce across the 16 sub-lanes of each half-wave
    for (int m = 8; m >= 1; m >>= 1)
#pragma unroll
      for (int v = 0; v < 8; ++v)
        ep[v] += __shfl_xor(ep[v], m, 32);
    if (sub == 0) {
#pragma unroll
      for (int v = 0; v < 8; ++v)
        energies[r0 + v + 8 * hi] = ep[v];
    }
  }
}

// ---------------------------------------------------------------------------
// Per-batch softmax over S_IN and context = weights @ enc_out; context (bf16)
// written into the second half of the decoder concat input buffer.
// ---------------------------------------------------------------------------
__global__ void softmax_ctx_kernel(const float* __restrict__ energies,
                                   const u16* __restrict__ X, u16* __restrict__ cat_bf) {
  __shared__ float w[S_IN];
  __shared__ float red[256];
  const int b = blockIdx.x, tid = threadIdx.x;
  const float* e = energies + (size_t)b * S_IN;

  float mx = -1e30f;
  for (int s = tid; s < S_IN; s += 256) mx = fmaxf(mx, e[s]);
  red[tid] = mx; __syncthreads();
  for (int k = 128; k >= 1; k >>= 1) { if (tid < k) red[tid] = fmaxf(red[tid], red[tid + k]); __syncthreads(); }
  mx = red[0]; __syncthreads();

  float sum = 0.f;
  for (int s = tid; s < S_IN; s += 256) { float ev = __expf(e[s] - mx); w[s] = ev; sum += ev; }
  red[tid] = sum; __syncthreads();
  for (int k = 128; k >= 1; k >>= 1) { if (tid < k) red[tid] += red[tid + k]; __syncthreads(); }
  float inv = 1.f / red[0];
  __syncthreads();

  const u16* xb = X + (size_t)b * S_IN * H;
  for (int h = tid; h < H; h += 256) {
    float acc = 0.f;
    for (int s = 0; s < S_IN; ++s) acc += w[s] * bf2f(xb[(size_t)s * H + h]);
    cat_bf[(size_t)b * (2 * H) + H + h] = f2bf(acc * inv);
  }
}

// ---------------------------------------------------------------------------
// Persistent decoder: 256 steps of
//   gh = h @ dec_W_hh^T (WMMA, NT=2 groups) -> sync -> gates -> h_gru into
//   cat_bf[:, :H] -> sync -> h = [h_gru, ctx] @ concat_W^T + b (WMMA) -> sync
// ---------------------------------------------------------------------------
__global__ void decoder_kernel(const float* __restrict__ x, const float* __restrict__ Wih,
                               const float* __restrict__ bih, const float* __restrict__ bhh,
                               const u16* __restrict__ Whh_bf, const u16* __restrict__ Wcat_bf,
                               const float* __restrict__ bcat, u16* h_bf, float* h_f,
                               float* gh, u16* cat_bf, u32* bar, int nblk) {
  const int tid = blockIdx.x * blockDim.x + threadIdx.x;
  const int nthreads = nblk * blockDim.x;
  const int wave = tid >> 5;
  const int nwaves = nthreads >> 5;
  const int NG1 = (B / 16) * (H3 / 32);   // 384
  const int NT2 = (B / 16) * (H / 16);    // 256

  for (int t = 0; t < S_OUT; ++t) {
    for (int g = wave; g < NG1; g += nwaves) {
      int tm = (g & 3) * 16;
      int tn = (g >> 2) * 32;
      v8f acc[2] = {{}, {}};
      wmma_dot_pipe<2, H / 32, 4>(h_bf, H, Whh_bf, H, tm, tn, acc);
      store_tile(gh, H3, tm, tn, acc[0], nullptr);
      store_tile(gh, H3, tm, tn + 16, acc[1], nullptr);
    }
    grid_sync(bar, nblk);

    for (int i = tid; i < B * H; i += nthreads) {
      int b = i >> 10, j = i & (H - 1);
      float xv = x[b * S_OUT + t];
      float ir = xv * Wih[j]          + bih[j];
      float iz = xv * Wih[H + j]      + bih[H + j];
      float in_ = xv * Wih[2 * H + j] + bih[2 * H + j];
      const float* g = gh + (size_t)b * H3;
      float gr = g[j]         + bhh[j];
      float gz = g[H + j]     + bhh[H + j];
      float gn = g[2 * H + j] + bhh[2 * H + j];
      float r = sigmoidf(ir + gr);
      float z = sigmoidf(iz + gz);
      float n = tanhf(in_ + r * gn);
      float hv = (1.f - z) * n + z * h_f[i];        // h_gru
      cat_bf[(size_t)b * (2 * H) + j] = f2bf(hv);   // first half of concat input
    }
    grid_sync(bar, nblk);

    for (int tile = wave; tile < NT2; tile += nwaves) {
      int tm = (tile & 3) * 16;
      int tn = (tile >> 2) * 16;
      v8f acc[1] = {{}};
      wmma_dot_pipe<1, (2 * H) / 32, 4>(cat_bf, 2 * H, Wcat_bf, 2 * H, tm, tn, acc);
      store_tile_dual(h_f, h_bf, H, tm, tn, acc[0], bcat);
    }
    grid_sync(bar, nblk);
  }
}

// ---------------------------------------------------------------------------
// out = last h_gru (bf16, in cat_bf[:, :H], lda=2H) @ out_W^T + out_b
// ---------------------------------------------------------------------------
__global__ void out_kernel(const u16* __restrict__ cat_bf, const u16* __restrict__ Wo,
                           const float* __restrict__ bo, float* __restrict__ out) {
  const int wave = (blockIdx.x * blockDim.x + threadIdx.x) >> 5;
  const int nwaves = (gridDim.x * blockDim.x) >> 5;
  const int NTILE = (B / 16) * (OUT_N / 16);  // 128
  for (int tile = wave; tile < NTILE; tile += nwaves) {
    int tm = (tile & 3) * 16;
    int tn = (tile >> 2) * 16;
    v8f acc[1] = {{}};
    wmma_dot_pipe<1, H / 32, 4>(cat_bf, 2 * H, Wo, H, tm, tn, acc);
    store_tile(out, OUT_N, tm, tn, acc[0], bo);
  }
}

// ---------------------------------------------------------------------------
extern "C" void kernel_launch(void* const* d_in, const int* in_sizes, int n_in,
                              void* d_out, int out_size, void* d_ws, size_t ws_size,
                              hipStream_t stream) {
  const float* input_ftrs  = (const float*)d_in[0];
  const float* output_ftrs = (const float*)d_in[1];
  const float* enc_W_ih = (const float*)d_in[2];
  const float* enc_W_hh = (const float*)d_in[3];
  const float* enc_b_ih = (const float*)d_in[4];
  const float* enc_b_hh = (const float*)d_in[5];
  const float* dec_W_ih = (const float*)d_in[6];
  const float* dec_W_hh = (const float*)d_in[7];
  const float* dec_b_ih = (const float*)d_in[8];
  const float* dec_b_hh = (const float*)d_in[9];
  const float* attn_W   = (const float*)d_in[10];
  const float* attn_b   = (const float*)d_in[11];
  const float* concat_W = (const float*)d_in[12];
  const float* concat_b = (const float*)d_in[13];
  const float* out_W    = (const float*)d_in[14];
  const float* out_b    = (const float*)d_in[15];
  float* out = (float*)d_out;

  char* ws = (char*)d_ws;
  size_t off = 0;
  auto alloc = [&](size_t bytes) -> void* {
    void* p = ws + off;
    off = (off + bytes + 255) & ~(size_t)255;
    return p;
  };
  u32*   bar      = (u32*)  alloc(256);
  float* gh       = (float*)alloc((size_t)B * H3 * 4);
  float* h_f      = (float*)alloc((size_t)B * H * 4);
  u16*   h_bf     = (u16*)  alloc((size_t)B * H * 2);
  u16*   cat_bf   = (u16*)  alloc((size_t)B * 2 * H * 2);
  float* energies = (float*)alloc((size_t)B * S_IN * 4);
  u16*   eWhh_bf  = (u16*)  alloc((size_t)H3 * H * 2);
  u16*   dWhh_bf  = (u16*)  alloc((size_t)H3 * H * 2);
  u16*   aW_bf    = (u16*)  alloc((size_t)H * H * 2);
  u16*   cW_bf    = (u16*)  alloc((size_t)H * 2 * H * 2);
  u16*   oW_bf    = (u16*)  alloc((size_t)OUT_N * H * 2);
  u16*   enc_bf   = (u16*)  alloc((size_t)B * S_IN * H * 2);

  cvt_kernel<<<256, 256, 0, stream>>>(enc_W_hh, eWhh_bf, H3 * H);
  cvt_kernel<<<256, 256, 0, stream>>>(dec_W_hh, dWhh_bf, H3 * H);
  cvt_kernel<<<64, 256, 0, stream>>>(attn_W, aW_bf, H * H);
  cvt_kernel<<<128, 256, 0, stream>>>(concat_W, cW_bf, H * 2 * H);
  cvt_kernel<<<64, 256, 0, stream>>>(out_W, oW_bf, OUT_N * H);
  init_kernel<<<64, 256, 0, stream>>>(bar, h_f, h_bf);

  const int NBLK = 48;   // 48 blocks x 8 waves = 384 waves == GEMM1 group count
  encoder_kernel<<<NBLK, 256, 0, stream>>>(input_ftrs, enc_W_ih, enc_b_ih, enc_b_hh,
                                           eWhh_bf, h_bf, h_f, gh, enc_bf, bar, NBLK);
  attn_energy_kernel<<<256, 256, 0, stream>>>(enc_bf, aW_bf, attn_b, energies);
  softmax_ctx_kernel<<<B, 256, 0, stream>>>(energies, enc_bf, cat_bf);
  decoder_kernel<<<NBLK, 256, 0, stream>>>(output_ftrs, dec_W_ih, dec_b_ih, dec_b_hh,
                                           dWhh_bf, cW_bf, concat_b, h_bf, h_f, gh,
                                           cat_bf, bar, NBLK);
  out_kernel<<<32, 256, 0, stream>>>(cat_bf, oW_bf, out_b, out);
}
